// Attention_86809878986979
// MI455X (gfx1250) — matrix operand order repaired and checked by reading, not compiled, not run on hardware
//
#include <hip/hip_runtime.h>
#include <hip/hip_bf16.h>
#include <stdint.h>

// Problem constants (from reference setup_inputs)
#define BATCH 16
#define TQ    2048
#define TS    2048
#define HD    1024

typedef __attribute__((ext_vector_type(16))) __bf16 v16bf;
typedef __attribute__((ext_vector_type(8)))  __bf16 v8bf;
typedef __attribute__((ext_vector_type(8)))  float  v8f;

static constexpr int BM  = 128;   // block tile M
static constexpr int BN  = 128;   // block tile N
static constexpr int BK  = 64;    // block tile K (two wmma k-steps of 32)
static constexpr int LDT = BK + 8; // padded LDS row stride in halves (144B: 16B-aligned, bank-conflict-free)

#if __has_builtin(__builtin_amdgcn_global_load_async_to_lds_b128)
#define ASYNC_LDS 1
// Builtin signature (from clang diagnostic): param 0 is
//   'int __attribute__((vector_size(16))) __device__ *'  (v4i in addrspace(1))
typedef int v4i_vs __attribute__((vector_size(16)));
typedef __attribute__((address_space(1))) v4i_vs *gas_t;  // global v4i*
typedef __attribute__((address_space(3))) v4i_vs *las_t;  // LDS v4i*
#endif

__device__ __forceinline__ v8f wmma_bf16(v16bf a, v16bf b, v8f c) {
  // D = A(16x32 bf16) * B(32x16 bf16) + C(16x16 f32)
  return __builtin_amdgcn_wmma_f32_16x16x32_bf16(false, a, false, b, (short)0, c,
                                                 false, false);
}

// A-matrix (16x32, 16-bit) fragment load from row-major LDS tile.
// Per ISA layout: lane m=lane&15, hi=lane>>4; VGPR0-3 hold k = hi*8+[0,8),
// VGPR4-7 hold k = 16+hi*8+[0,8).  Both are contiguous 16B chunks of row m.
__device__ __forceinline__ v16bf ld_frag_a(const __bf16* base, int lane) {
  const int m  = lane & 15;
  const int hi = (lane >> 4) & 1;
  const __bf16* r = base + m * LDT;
  v8bf lo = *(const v8bf*)(r + hi * 8);
  v8bf hb = *(const v8bf*)(r + 16 + hi * 8);
  v16bf a;
#pragma unroll
  for (int i = 0; i < 8; ++i) { a[i] = lo[i]; a[i + 8] = hb[i]; }
  return a;
}

// B-matrix (32x16, 16-bit) fragment load from an LDS tile stored [n][k].
// Per ISA layout: lane n=lane&15, hi=lane>>4; VGPR0-3 hold k = hi*8+[0,8),
// VGPR4-7 hold k = 16+hi*8+[0,8).  Both are contiguous 16B chunks of row n.
__device__ __forceinline__ v16bf ld_frag_b(const __bf16* base, int lane) {
  const int n  = lane & 15;
  const int hi = (lane >> 4) & 1;
  const __bf16* r = base + n * LDT;
  v8bf lo = *(const v8bf*)(r + hi * 8);
  v8bf hb = *(const v8bf*)(r + 16 + hi * 8);
  v16bf b;
#pragma unroll
  for (int i = 0; i < 8; ++i) { b[i] = lo[i]; b[i + 8] = hb[i]; }
  return b;
}

// ---------------------------------------------------------------------------
// Kernel 1: S[b][q][s] = sum_h Q[b][q][h] * K[b][s][h]   (NT GEMM, bf16 wmma)
// Grid: (TS/BN, TQ/BM, BATCH), 256 threads (8 waves; wave tile 64x32)
// ---------------------------------------------------------------------------
__global__ __launch_bounds__(256) void qk_scores_kernel(
    const float* __restrict__ Q, const float* __restrict__ K,
    float* __restrict__ S) {
  __shared__ __attribute__((aligned(16))) __bf16 sQ[BM * LDT];
  __shared__ __attribute__((aligned(16))) __bf16 sK[BN * LDT];

  const int b  = blockIdx.z;
  const int qt = blockIdx.y;
  const int st = blockIdx.x;
  const float* Qb = Q + ((size_t)b * TQ + (size_t)qt * BM) * HD;
  const float* Kb = K + ((size_t)b * TS + (size_t)st * BN) * HD;
  float*       Sb = S + ((size_t)b * TQ + (size_t)qt * BM) * TS + (size_t)st * BN;

  const int tid  = threadIdx.x;
  const int wave = tid >> 5;
  const int lane = tid & 31;
  const int wm = (wave >> 2) * 64;  // wave q-offset: 0 / 64
  const int wn = (wave & 3) * 32;   // wave s-offset: 0..96

  v8f acc[4][2] = {};

  const int row  = tid >> 1;        // 0..127 (tile row)
  const int half = tid & 1;         // which 32-float half of the row

  for (int k0 = 0; k0 < HD; k0 += BK) {
    // Stage Q and K tiles (f32 -> bf16 convert) into LDS, row-major [row][h].
    {
      const float4* gq = (const float4*)(Qb + (size_t)row * HD + k0 + half * 32);
      const float4* gk = (const float4*)(Kb + (size_t)row * HD + k0 + half * 32);
      if (k0 + BK < HD) {
        __builtin_prefetch(Qb + (size_t)row * HD + k0 + BK, 0, 0);
        __builtin_prefetch(Kb + (size_t)row * HD + k0 + BK, 0, 0);
      }
      __bf16* dq = sQ + row * LDT + half * 32;
      __bf16* dk = sK + row * LDT + half * 32;
#pragma unroll
      for (int i = 0; i < 8; ++i) {
        float4 v = gq[i];
        dq[4 * i + 0] = (__bf16)v.x; dq[4 * i + 1] = (__bf16)v.y;
        dq[4 * i + 2] = (__bf16)v.z; dq[4 * i + 3] = (__bf16)v.w;
        float4 w = gk[i];
        dk[4 * i + 0] = (__bf16)w.x; dk[4 * i + 1] = (__bf16)w.y;
        dk[4 * i + 2] = (__bf16)w.z; dk[4 * i + 3] = (__bf16)w.w;
      }
    }
    __syncthreads();

#pragma unroll
    for (int kk = 0; kk < BK; kk += 32) {
      v16bf af[4];
#pragma unroll
      for (int mi = 0; mi < 4; ++mi)
        af[mi] = ld_frag_a(sQ + (wm + mi * 16) * LDT + kk, lane);
      v16bf bf_[2];
#pragma unroll
      for (int ni = 0; ni < 2; ++ni)
        bf_[ni] = ld_frag_b(sK + (wn + ni * 16) * LDT + kk, lane);
#pragma unroll
      for (int mi = 0; mi < 4; ++mi)
#pragma unroll
        for (int ni = 0; ni < 2; ++ni)
          acc[mi][ni] = wmma_bf16(af[mi], bf_[ni], acc[mi][ni]);
    }
    __syncthreads();
  }

  // C layout: lane<16 -> (M=r, N=lane); lane>=16 -> (M=8+r, N=lane-16)
  const int n  = lane & 15;
  const int mh = (lane >> 4) * 8;
#pragma unroll
  for (int mi = 0; mi < 4; ++mi)
#pragma unroll
    for (int ni = 0; ni < 2; ++ni)
#pragma unroll
      for (int r = 0; r < 8; ++r)
        Sb[(size_t)(wm + mi * 16 + mh + r) * TS + (wn + ni * 16 + n)] =
            acc[mi][ni][r];
}

// ---------------------------------------------------------------------------
// Kernel 2: row softmax over s (contiguous), writing bf16 probabilities
// IN PLACE into the front half of each fp32 row.  One WG per (b,q) row.
// ---------------------------------------------------------------------------
__global__ __launch_bounds__(256) void softmax_rows_kernel(float* __restrict__ S) {
  float*   row  = S + (size_t)blockIdx.x * TS;
  __bf16*  prow = (__bf16*)row;  // bf16 row aliases first half of f32 row
  const int tid = threadIdx.x;

  const float4* r4 = (const float4*)row;
  float4 a = r4[tid * 2];
  float4 c = r4[tid * 2 + 1];
  float x[8] = {a.x, a.y, a.z, a.w, c.x, c.y, c.z, c.w};

  float m = x[0];
#pragma unroll
  for (int i = 1; i < 8; ++i) m = fmaxf(m, x[i]);

  __shared__ float red[256];
  red[tid] = m;
  __syncthreads();
  for (int s = 128; s > 0; s >>= 1) {
    if (tid < s) red[tid] = fmaxf(red[tid], red[tid + s]);
    __syncthreads();
  }
  const float rowmax = red[0];
  __syncthreads();

  float e[8];
  float sum = 0.f;
#pragma unroll
  for (int i = 0; i < 8; ++i) { e[i] = __expf(x[i] - rowmax); sum += e[i]; }
  red[tid] = sum;
  __syncthreads();
  for (int s = 128; s > 0; s >>= 1) {
    if (tid < s) red[tid] += red[tid + s];
    __syncthreads();
  }
  const float inv = 1.f / red[0];

  v8bf o;
#pragma unroll
  for (int i = 0; i < 8; ++i) o[i] = (__bf16)(e[i] * inv);
  *(v8bf*)(prow + tid * 8) = o;  // all reads of this row completed before barriers
}

// ---------------------------------------------------------------------------
// Kernel 3: ctx[b][q][h] = sum_s P[b][q][s] * V[b][s][h]
// P is bf16 rows embedded in the f32 score buffer (row stride TS f32 = 4096 bf16).
// V is staged TRANSPOSED ([h][s]) into LDS so B fragments are k-contiguous.
// Grid: (HD/BN, TQ/BM, BATCH), 256 threads.
// ---------------------------------------------------------------------------
__global__ __launch_bounds__(256) void pv_context_kernel(
    const float* __restrict__ V, const float* __restrict__ Sws,
    float* __restrict__ out) {
  __shared__ __attribute__((aligned(16))) __bf16 sP[BM * LDT];  // [q][s]
  __shared__ __attribute__((aligned(16))) __bf16 sV[BN * LDT];  // [h][s] (transposed)

  const int b  = blockIdx.z;
  const int qt = blockIdx.y;
  const int ht = blockIdx.x;
  const float* Vb   = V + (size_t)b * TS * HD;
  float*       outb = out + ((size_t)b * TQ + (size_t)qt * BM) * HD + (size_t)ht * BN;

  const int tid  = threadIdx.x;
  const int wave = tid >> 5;
  const int lane = tid & 31;
  const int wm = (wave >> 2) * 64;  // q offset
  const int wn = (wave & 3) * 32;   // h offset

  v8f acc[4][2] = {};

  const int prow_i = tid >> 1;      // 0..127 q-row for P staging
  const int phalf  = tid & 1;
  const int srow   = tid >> 2;      // 0..63 s-row for V staging
  const int q4     = tid & 3;       // which 32-wide h chunk

  for (int s0 = 0; s0 < TS; s0 += BK) {
    // ---- stage P tile [128 q][64 s] bf16 (pure copy -> async LDS DMA) ----
    {
      const __bf16* pg =
          (const __bf16*)(Sws + ((size_t)b * TQ + (size_t)qt * BM + prow_i) * TS) +
          s0 + phalf * 32;
      __bf16* pl = sP + prow_i * LDT + phalf * 32;
#if defined(ASYNC_LDS)
      const char* gc = (const char*)pg;
      char*       lc = (char*)pl;
#pragma unroll
      for (int i = 0; i < 4; ++i)
        __builtin_amdgcn_global_load_async_to_lds_b128(
            (gas_t)(uintptr_t)(gc + i * 16),
            (las_t)(uint32_t)(uintptr_t)(lc + i * 16), 0, 0);
#else
      const uint4* gp4 = (const uint4*)pg;
      uint4*       lp4 = (uint4*)pl;
#pragma unroll
      for (int i = 0; i < 4; ++i) lp4[i] = gp4[i];
#endif
    }
    // ---- stage V tile [64 s][128 h] f32 -> bf16, transposed to [h][s] ----
    {
      const float4* gv =
          (const float4*)(Vb + (size_t)(s0 + srow) * HD + ht * BN + q4 * 32);
#pragma unroll
      for (int i = 0; i < 8; ++i) {
        float4 w = gv[i];
        const int h = q4 * 32 + i * 4;
        sV[(h + 0) * LDT + srow] = (__bf16)w.x;
        sV[(h + 1) * LDT + srow] = (__bf16)w.y;
        sV[(h + 2) * LDT + srow] = (__bf16)w.z;
        sV[(h + 3) * LDT + srow] = (__bf16)w.w;
      }
    }
#if defined(ASYNC_LDS)
#if __has_builtin(__builtin_amdgcn_s_wait_asynccnt)
    __builtin_amdgcn_s_wait_asynccnt(0);
#else
    asm volatile("s_wait_asynccnt 0" ::: "memory");
#endif
#endif
    __syncthreads();

#pragma unroll
    for (int kk = 0; kk < BK; kk += 32) {
      v16bf af[4];
#pragma unroll
      for (int mi = 0; mi < 4; ++mi)
        af[mi] = ld_frag_a(sP + (wm + mi * 16) * LDT + kk, lane);
      v16bf bf_[2];
#pragma unroll
      for (int ni = 0; ni < 2; ++ni)
        bf_[ni] = ld_frag_b(sV + (wn + ni * 16) * LDT + kk, lane);
#pragma unroll
      for (int mi = 0; mi < 4; ++mi)
#pragma unroll
        for (int ni = 0; ni < 2; ++ni)
          acc[mi][ni] = wmma_bf16(af[mi], bf_[ni], acc[mi][ni]);
    }
    __syncthreads();
  }

  const int n  = lane & 15;
  const int mh = (lane >> 4) * 8;
#pragma unroll
  for (int mi = 0; mi < 4; ++mi)
#pragma unroll
    for (int ni = 0; ni < 2; ++ni)
#pragma unroll
      for (int r = 0; r < 8; ++r)
        outb[(size_t)(wm + mi * 16 + mh + r) * HD + (wn + ni * 16 + n)] =
            acc[mi][ni][r];
}

// ---------------------------------------------------------------------------
// Launch: QK^T scores -> softmax over s -> P*V context.
// Workspace: needs BATCH*TQ*TS*4 = 256 MiB (fp32 scores; bf16 probabilities
// are written in place into the same buffer).
// ---------------------------------------------------------------------------
extern "C" void kernel_launch(void* const* d_in, const int* in_sizes, int n_in,
                              void* d_out, int out_size, void* d_ws, size_t ws_size,
                              hipStream_t stream) {
  (void)in_sizes; (void)n_in; (void)out_size; (void)ws_size;
  const float* hidden = (const float*)d_in[0];  // [B, Tq, H]
  const float* enc    = (const float*)d_in[1];  // [B, Ts, H]
  float*       out    = (float*)d_out;          // [B, Tq, H]
  float*       S      = (float*)d_ws;           // [B, Tq, Ts] fp32 scratch

  dim3 gridA(TS / BN, TQ / BM, BATCH);
  qk_scores_kernel<<<gridA, 256, 0, stream>>>(hidden, enc, S);

  softmax_rows_kernel<<<BATCH * TQ, 256, 0, stream>>>(S);

  dim3 gridC(HD / BN, TQ / BM, BATCH);
  pv_context_kernel<<<gridC, 256, 0, stream>>>(enc, S, out);
}
